// MultiHeadAttention_87643102642593
// MI455X (gfx1250) — matrix operand order repaired and checked by reading, not compiled, not run on hardware
//
#include <hip/hip_runtime.h>
#include <hip/hip_bf16.h>

// ---------------------------------------------------------------------------
// MHA forward for MI455X (gfx1250, wave32, WMMA).
// B=4 S=2048 E=1024 H=16 D=64. All matmuls via v_wmma_f32_16x16x32_bf16.
// Phase 1: Q/K/V projections (fp32 in -> bf16 out, fp32 accumulate).
// Phase 2: flash-attention per (b, h, 64-query tile), online softmax fp32.
//
// All WMMA fragments are fetched from LDS with 16-byte ds_load_b128 pairs:
//   A frag : row-major tile, stride 72 ushorts (144 B = 16 B aligned rows);
//            per lane two contiguous 8-ushort runs.
//   B frag : tile stored [N][K] ("transposed"), per lane two 8-ushort runs
//            (K and K+16), each 16 B aligned.
// ---------------------------------------------------------------------------

typedef __attribute__((ext_vector_type(16))) __bf16       v16bf;
typedef __attribute__((ext_vector_type(8)))  float        v8f;
typedef __attribute__((ext_vector_type(4)))  unsigned int v4u;

union FragB16 {
    v4u   q[2];
    v16bf v;
};

#define LD 72   // LDS row stride in ushorts: 144 B, multiple of 16 B

__device__ static inline unsigned short f2bf(float f) {
    union { float f; unsigned int u; } x; x.f = f;
    unsigned int r = x.u + 0x7FFFu + ((x.u >> 16) & 1u);   // round-to-nearest-even
    return (unsigned short)(r >> 16);
}

__device__ static inline unsigned int f2bf_pk(float lo, float hi) {
    return (unsigned int)f2bf(lo) | ((unsigned int)f2bf(hi) << 16);
}

__device__ static inline v8f wmma_bf16(v16bf a, v16bf b, v8f c) {
    // (neg_a, A, neg_b, B, c_mod, C, reuse_a, reuse_b)
    return __builtin_amdgcn_wmma_f32_16x16x32_bf16(false, a, false, b, (short)0, c,
                                                   false, false);
}

// A-fragment: 16x32 bf16 from row-major [M][K] LDS tile (stride LD).
// ISA layout: lane<16 -> M=lane, VGPR0-3 K=0..7(+8*half), VGPR4-7 K=16..23(+8*half).
// Two contiguous 16 B runs per lane -> 2x ds_load_b128.
__device__ static inline v16bf load_a_frag(const unsigned short* lds,
                                           int row0, int k0) {
    const int lane = threadIdx.x & 31;
    const int half = lane >> 4;
    const int m    = lane & 15;
    const unsigned short* base = lds + (row0 + m) * LD + k0 + 8 * half;
    FragB16 f;
    f.q[0] = *(const v4u*)(base);        // K = 0..7   (+8*half)
    f.q[1] = *(const v4u*)(base + 16);   // K = 16..23 (+8*half)
    return f.v;
}

// B-fragment: 32x16 bf16 where the LDS tile is stored [N][K] (stride LD).
// Lane layout: N=lane&15, K = 0..7 and 16..23 (+8*(lane>>4)): two 8-ushort runs
// per lane -> 2x ds_load_b128.
__device__ static inline v16bf load_b_frag(const unsigned short* lds,
                                           int k0, int n0) {
    const int lane  = threadIdx.x & 31;
    const int khalf = lane >> 4;
    const int n     = lane & 15;
    const unsigned short* base = lds + (n0 + n) * LD + k0 + 8 * khalf;
    FragB16 f;
    f.q[0] = *(const v4u*)(base);        // K = 0..7   (+8*half)
    f.q[1] = *(const v4u*)(base + 16);   // K = 16..23 (+8*half)
    return f.v;
}

// ---------------------------------------------------------------------------
// Phase 1: Out(bf16, 8192x1024) = X(fp32, 8192x1024) * W(fp32, 1024x1024)
// Block: 128(M) x 64(N), 256 threads = 8 waves (4 along M x 2 along N),
// each wave 32x32 via 2x2 WMMA tiles. K staged 64 wide through LDS.
// W tile staged transposed ([n][k]) so B fragments are ds_load_b128.
// ---------------------------------------------------------------------------
__global__ __launch_bounds__(256) void qkv_proj_kernel(
    const float* __restrict__ X, const float* __restrict__ W,
    unsigned short* __restrict__ Out)
{
    __shared__ unsigned short a_lds[128 * LD];   // [m][k]
    __shared__ unsigned short w_lds[64 * LD];    // [n][k]  (transposed W tile)

    const int tid  = threadIdx.x;
    const int lane = tid & 31;
    const int wave = tid >> 5;
    const int wm   = wave & 3;            // wave position along M (x32)
    const int wn   = wave >> 2;           // wave position along N (x32)
    const int m0   = blockIdx.x * 128;
    const int n0   = blockIdx.y * 64;

    v8f acc[2][2] = {};

    for (int kc = 0; kc < 16; ++kc) {
        const int k0 = kc * 64;
        // Stage A tile 128x64 (fp32 -> packed bf16, 32-bit LDS stores).
        for (int i = tid; i < 128 * 32; i += 256) {
            const int r  = i >> 5;
            const int c2 = (i & 31) * 2;
            const float x0 = X[(size_t)(m0 + r) * 1024 + k0 + c2];
            const float x1 = X[(size_t)(m0 + r) * 1024 + k0 + c2 + 1];
            *(unsigned int*)&a_lds[r * LD + c2] = f2bf_pk(x0, x1);
        }
        // Stage W tile 64x64 transposed: read coalesced along n, store [n][k].
        for (int i = tid; i < 64 * 64; i += 256) {
            const int r = i >> 6, c = i & 63;   // r = k, c = n
            w_lds[c * LD + r] = f2bf(W[(size_t)(k0 + r) * 1024 + n0 + c]);
        }
        if (kc + 1 < 16) {  // gfx1250 global_prefetch_b8 for next K chunk
            __builtin_prefetch(&X[(size_t)(m0 + (tid >> 1)) * 1024 + k0 + 64], 0, 1);
            __builtin_prefetch(&W[(size_t)(k0 + 64 + (tid >> 2)) * 1024 + n0], 0, 1);
        }
        __syncthreads();

#pragma unroll
        for (int ks = 0; ks < 2; ++ks) {
            const v16bf a0 = load_a_frag(a_lds, wm * 32 + 0,  ks * 32);
            const v16bf a1 = load_a_frag(a_lds, wm * 32 + 16, ks * 32);
            const v16bf b0 = load_b_frag(w_lds, ks * 32, wn * 32 + 0);
            const v16bf b1 = load_b_frag(w_lds, ks * 32, wn * 32 + 16);
            acc[0][0] = wmma_bf16(a0, b0, acc[0][0]);
            acc[0][1] = wmma_bf16(a0, b1, acc[0][1]);
            acc[1][0] = wmma_bf16(a1, b0, acc[1][0]);
            acc[1][1] = wmma_bf16(a1, b1, acc[1][1]);
        }
        __syncthreads();
    }

    // C layout: VGPR r -> (M = r + 8*half, N = lane&15) within each 16x16 tile.
    const int half = lane >> 4;
    const int nl   = lane & 15;
#pragma unroll
    for (int i = 0; i < 2; ++i)
#pragma unroll
        for (int j = 0; j < 2; ++j) {
            const int rowb = m0 + wm * 32 + i * 16 + 8 * half;
            const int col  = n0 + wn * 32 + j * 16 + nl;
#pragma unroll
            for (int r = 0; r < 8; ++r)
                Out[(size_t)(rowb + r) * 1024 + col] = f2bf(acc[i][j][r]);
        }
}

// ---------------------------------------------------------------------------
// Phase 2: flash attention. Grid (S/64, H, B), 128 threads = 4 waves.
// Each wave owns 16 query rows; block processes 64 queries of one head,
// streaming 64-key chunks: scores (WMMA) -> online softmax (fp32 LDS) ->
// P*V accumulate (WMMA, O in registers) -> normalize by running sum.
// K tile kept [key][d] (= [N][K] for scores); V tile staged transposed
// [d][key] (= [N][K] for the P*V product).
// ---------------------------------------------------------------------------
__global__ __launch_bounds__(128) void attn_kernel(
    const unsigned short* __restrict__ Qb,
    const unsigned short* __restrict__ Kb,
    const unsigned short* __restrict__ Vb,
    float* __restrict__ Out)
{
    __shared__ unsigned short q_lds[64 * LD];   // [q][d]
    __shared__ unsigned short k_lds[64 * LD];   // [key][d]
    __shared__ unsigned short v_lds[64 * LD];   // [d][key] (transposed)
    __shared__ float          s_lds[64 * 65];   // scores, stride 65 for banks
    __shared__ unsigned short p_lds[64 * LD];   // [q][key]
    __shared__ float m_s[64], l_s[64], f_s[64];

    const int tid  = threadIdx.x;
    const int lane = tid & 31;
    const int wave = tid >> 5;           // wave w -> query rows 16w..16w+15
    const int q0   = blockIdx.x * 64;
    const int h    = blockIdx.y;
    const int b    = blockIdx.z;

    const size_t hoff = (size_t)h * 64;
    const size_t bso  = (size_t)b * 2048;

    // Stage Q tile once (64 rows x 64 dims, bf16, 32-bit copies).
    for (int i = tid; i < 64 * 32; i += 128) {
        const int r  = i >> 5;
        const int c2 = (i & 31) * 2;
        *(unsigned int*)&q_lds[r * LD + c2] =
            *(const unsigned int*)&Qb[(bso + q0 + r) * 1024 + hoff + c2];
    }
    if (tid < 64) { m_s[tid] = -3.0e38f; l_s[tid] = 0.0f; }
    __syncthreads();

    const int half  = lane >> 4;
    const int nl    = lane & 15;
    const int rbase = wave * 16 + 8 * half;   // absolute row of acc element r

    v8f oacc[4] = {};                         // 16 rows x 64 dims per wave

    for (int kc = 0; kc < 32; ++kc) {
        const int s0 = kc * 64;
        // Stage K chunk (32-bit copies) and V chunk (transposed to [d][key]).
        for (int i = tid; i < 64 * 32; i += 128) {
            const int r  = i >> 5;
            const int c2 = (i & 31) * 2;
            const size_t g = (bso + s0 + r) * 1024 + hoff + c2;
            *(unsigned int*)&k_lds[r * LD + c2] = *(const unsigned int*)&Kb[g];
            const unsigned int vv = *(const unsigned int*)&Vb[g];
            v_lds[(c2 + 0) * LD + r] = (unsigned short)(vv & 0xFFFFu);
            v_lds[(c2 + 1) * LD + r] = (unsigned short)(vv >> 16);
        }
        if (kc + 1 < 32) {
            const size_t g = (bso + s0 + 64 + (tid & 63)) * 1024 + hoff;
            __builtin_prefetch(&Kb[g], 0, 1);
            __builtin_prefetch(&Vb[g], 0, 1);
        }
        __syncthreads();

        // ---- scores: S(16x64) = Q(16x64) . K^T, scaled by 1/sqrt(64) ----
        {
            const v16bf a0 = load_a_frag(q_lds, wave * 16, 0);
            const v16bf a1 = load_a_frag(q_lds, wave * 16, 32);
#pragma unroll
            for (int t = 0; t < 4; ++t) {
                v8f s = {};
                s = wmma_bf16(a0, load_b_frag(k_lds, 0,  t * 16), s);
                s = wmma_bf16(a1, load_b_frag(k_lds, 32, t * 16), s);
#pragma unroll
                for (int r = 0; r < 8; ++r)
                    s_lds[(rbase + r) * 65 + t * 16 + nl] = s[r] * 0.125f;
            }
        }
        __syncthreads();

        // ---- online softmax: one thread per query row ----
        if (tid < 64) {
            const float mold = m_s[tid];
            const float lold = l_s[tid];
            float mx = -3.0e38f;
            for (int j = 0; j < 64; ++j) mx = fmaxf(mx, s_lds[tid * 65 + j]);
            const float mnew = fmaxf(mold, mx);
            const float fr   = __expf(mold - mnew);
            float sum = 0.0f;
            for (int j = 0; j < 64; j += 2) {
                const float p0 = __expf(s_lds[tid * 65 + j]     - mnew);
                const float p1 = __expf(s_lds[tid * 65 + j + 1] - mnew);
                sum += p0 + p1;
                *(unsigned int*)&p_lds[tid * LD + j] = f2bf_pk(p0, p1);
            }
            m_s[tid] = mnew;
            l_s[tid] = lold * fr + sum;
            f_s[tid] = fr;
        }
        __syncthreads();

        // ---- rescale O by exp(m_old - m_new), then O += P . V ----
#pragma unroll
        for (int r = 0; r < 8; ++r) {
            const float fr = f_s[rbase + r];
#pragma unroll
            for (int t = 0; t < 4; ++t) oacc[t][r] *= fr;
        }
        {
            const v16bf a0 = load_a_frag(p_lds, wave * 16, 0);
            const v16bf a1 = load_a_frag(p_lds, wave * 16, 32);
#pragma unroll
            for (int t = 0; t < 4; ++t) {
                oacc[t] = wmma_bf16(a0, load_b_frag(v_lds, 0,  t * 16), oacc[t]);
                oacc[t] = wmma_bf16(a1, load_b_frag(v_lds, 32, t * 16), oacc[t]);
            }
        }
        __syncthreads();   // before next chunk overwrites K/V/P
    }

    // ---- epilogue: O / l, write fp32 output [b][s][h*64+d] ----
#pragma unroll
    for (int r = 0; r < 8; ++r) {
        const float inv = 1.0f / l_s[rbase + r];
        const int   row = q0 + rbase + r;
#pragma unroll
        for (int t = 0; t < 4; ++t)
            Out[(bso + row) * 1024 + hoff + t * 16 + nl] = oacc[t][r] * inv;
    }
}

// ---------------------------------------------------------------------------
extern "C" void kernel_launch(void* const* d_in, const int* in_sizes, int n_in,
                              void* d_out, int out_size, void* d_ws, size_t ws_size,
                              hipStream_t stream) {
    (void)in_sizes; (void)n_in; (void)out_size; (void)ws_size;
    const float* q_in = (const float*)d_in[0];
    const float* k_in = (const float*)d_in[1];
    const float* v_in = (const float*)d_in[2];
    const float* Wq   = (const float*)d_in[3];
    const float* Wk   = (const float*)d_in[4];
    const float* Wv   = (const float*)d_in[5];

    // Workspace: Q/K/V projected to bf16, each 8192x1024 (16 MB) -> 48 MB total.
    unsigned short* qb = (unsigned short*)d_ws;
    unsigned short* kb = qb + (size_t)8192 * 1024;
    unsigned short* vb = kb + (size_t)8192 * 1024;

    dim3 pgrid(64, 16);          // M/128, N/64
    qkv_proj_kernel<<<pgrid, 256, 0, stream>>>(q_in, Wq, qb);
    qkv_proj_kernel<<<pgrid, 256, 0, stream>>>(k_in, Wk, kb);
    qkv_proj_kernel<<<pgrid, 256, 0, stream>>>(v_in, Wv, vb);

    dim3 agrid(32, 16, 4);       // S/64 query tiles, H, B
    attn_kernel<<<agrid, 128, 0, stream>>>(qb, kb, vb, (float*)d_out);
}
